// joint_actor_28690381537986
// MI455X (gfx1250) — matrix operand order repaired and checked by reading, not compiled, not run on hardware
//
#include <hip/hip_runtime.h>
#include <hip/hip_bf16.h>

typedef _Float16 h2   __attribute__((ext_vector_type(2)));
typedef _Float16 v16h __attribute__((ext_vector_type(16)));
typedef float    v8f  __attribute__((ext_vector_type(8)));

#define BQ      32768        // graphs per batch
#define NPG     25           // nodes per graph
#define GPB     16           // graphs per block
#define MROWS   (GPB*NPG)    // 400 rows per block
#define MTILES  (MROWS/16)   // 25 row tiles
#define HD      64
#define THREADS 512          // 16 waves: 4 col-tiles x 4 row-parities
#define HBYTES  (MROWS*HD*2) // one h buffer: 51.2 KB

#if __has_builtin(__builtin_amdgcn_tanhf)
#define FAST_TANH(x) __builtin_amdgcn_tanhf(x)     // native V_TANH_F32
#else
#define FAST_TANH(x) tanhf(x)
#endif

struct alignas(16) H8 { h2 q[4]; };          // 8 contiguous f16 in LDS
union  AFrag { v16h h; h2 p[8]; };           // one WMMA 16x32 f16 fragment

__device__ __forceinline__ void load8(h2* dst, const _Float16* p) {
    const H8 v = *reinterpret_cast<const H8*>(p);
    dst[0] = v.q[0]; dst[1] = v.q[1]; dst[2] = v.q[2]; dst[3] = v.q[3];
}

// Build B fragment (32x16, f16) for column `col`, k-step ks, from fp32 W[64][64]
__device__ __forceinline__ v16h loadBfrag(const float* __restrict__ W, int ks, int hi, int col) {
    AFrag b;
#pragma unroll
    for (int v = 0; v < 8; ++v) {
        const int k0 = 32*ks + 8*hi + 2*v + (v >= 4 ? 8 : 0);
        h2 e;
        e[0] = (_Float16)W[(k0    ) * 64 + col];
        e[1] = (_Float16)W[(k0 + 1) * 64 + col];
        b.p[v] = e;
    }
    return b.h;
}

__device__ __forceinline__ v8f wmma_f16(v16h a, v16h b, v8f c) {
    return __builtin_amdgcn_wmma_f32_16x16x32_f16(false, a, false, b, (short)0, c, false, false);
}

__global__ void __launch_bounds__(THREADS)
joint_actor_fused_kernel(const float* __restrict__ x,
                         const float* __restrict__ W_torso, const float* __restrict__ b_torso,
                         const float* __restrict__ W_joint, const float* __restrict__ b_joint,
                         const float* __restrict__ Wrel,    const float* __restrict__ brel,
                         const float* __restrict__ Wroot,
                         const float* __restrict__ Wj,      const float* __restrict__ bj,
                         const float* __restrict__ Wt,      const float* __restrict__ bt,
                         float* __restrict__ out)
{
    // Dynamic LDS: two ping-pong h buffers, 2 x 400 x 64 f16 = 102.4 KB (< 320 KB/WGP)
    extern __shared__ __align__(16) _Float16 smem[];
    _Float16* bufA = smem;
    _Float16* bufB = smem + MROWS * HD;

    const int tid  = threadIdx.x;
    const int lane = tid & 31;
    const int wave = tid >> 5;
    const int base_node = blockIdx.x * MROWS;

    // ---------------- Encoder -> bufA: two f16 outputs per iteration, ds_store_b32 ---------
    for (int idx = tid; idx < MROWS * 32; idx += THREADS) {
        const int r  = idx >> 5;
        const int kp = (idx & 31) * 2;
        const float* xp = x + (size_t)(base_node + r) * 4;
        float v0, v1;
        if ((r % 5) == 0) {  // torso positions 0,5,10,15,20
            v0 = b_torso[kp]   + xp[0]*W_torso[kp]     + xp[1]*W_torso[64+kp]
                               + xp[2]*W_torso[128+kp]   + xp[3]*W_torso[192+kp];
            v1 = b_torso[kp+1] + xp[0]*W_torso[kp+1]   + xp[1]*W_torso[64+kp+1]
                               + xp[2]*W_torso[128+kp+1] + xp[3]*W_torso[192+kp+1];
        } else {
            v0 = b_joint[kp]   + xp[0]*W_joint[kp]     + xp[1]*W_joint[64+kp];
            v1 = b_joint[kp+1] + xp[0]*W_joint[kp+1]   + xp[1]*W_joint[64+kp+1];
        }
        h2 pv; pv[0] = (_Float16)v0; pv[1] = (_Float16)v1;
        *reinterpret_cast<h2*>(&bufA[r * HD + kp]) = pv;
    }
    __syncthreads();

    // ---------------- 3 x (GraphConv + tanh): read cur, write nxt, swap --------------------
    const int ct      = wave & 3;       // column tile 0..3
    const int r0      = wave >> 2;      // row-tile parity 0..3
    const int colbase = 16 * ct;
    const int nlo     = lane & 15;
    const int hi      = lane >> 4;

    const _Float16* cur = bufA;
    _Float16*       nxt = bufB;

    for (int layer = 0; layer < 3; ++layer) {
        const float* WR = Wrel  + layer * 4096;
        const float* WO = Wroot + layer * 4096;
        const float  bias = brel[layer * 64 + colbase + nlo];
        if (layer < 2) {                      // warm next layer's weights into L2
            __builtin_prefetch(WR + 4096, 0, 1);
            __builtin_prefetch(WO + 4096, 0, 1);
        }

        // B fragments held in VGPRs for the whole layer (reused across 6-7 row tiles)
        const v16h bR0 = loadBfrag(WR, 0, hi, colbase + nlo);
        const v16h bR1 = loadBfrag(WR, 1, hi, colbase + nlo);
        const v16h bO0 = loadBfrag(WO, 0, hi, colbase + nlo);
        const v16h bO1 = loadBfrag(WO, 1, hi, colbase + nlo);

#pragma unroll
        for (int t = 0; t < 7; ++t) {
            const int rt = r0 + 4 * t;
            if (rt < MTILES) {
                const int row = 16 * rt + nlo;      // this lane's A row
                const int pos = row % NPG;          // position in the 25-node chain
                // branchless chain masks: clamped addresses + 0/1 f16 masks (no exec churn)
                const int rowm = (pos > 0)       ? row - 1 : row;
                const int rowp = (pos < NPG - 1) ? row + 1 : row;
                h2 vmp, vmn;
                vmp[0] = vmp[1] = (_Float16)((pos > 0)       ? 1.f : 0.f);
                vmn[0] = vmn[1] = (_Float16)((pos < NPG - 1) ? 1.f : 0.f);
                v8f c = {};
#pragma unroll
                for (int ks = 0; ks < 2; ++ks) {
                    const int kb = 32 * ks + 8 * hi;
                    AFrag ah, ap, an, ag;
                    load8(ah.p,     cur + row  * HD + kb);
                    load8(ah.p + 4, cur + row  * HD + kb + 16);
                    load8(ap.p,     cur + rowm * HD + kb);
                    load8(ap.p + 4, cur + rowm * HD + kb + 16);
                    load8(an.p,     cur + rowp * HD + kb);
                    load8(an.p + 4, cur + rowp * HD + kb + 16);
#pragma unroll
                    for (int j = 0; j < 8; ++j)      // v_pk_mul + v_pk_fma, straight-line
                        ag.p[j] = ap.p[j] * vmp + an.p[j] * vmn;
                    const v16h bR = ks ? bR1 : bR0;
                    const v16h bO = ks ? bO1 : bO0;
                    c = wmma_f16(ag.h, bR, c);       // agg @ Wrel  (shared f32 accumulator)
                    c = wmma_f16(ah.h, bO, c);       // h   @ Wroot
                }
                // epilogue straight to the ping-pong buffer: no long-lived accumulators
#pragma unroll
                for (int j = 0; j < 8; ++j) {
                    const int m = j + 8 * hi;        // D layout: lanes>=16 hold M=8..15
                    nxt[(16 * rt + m) * HD + colbase + nlo] =
                        (_Float16)FAST_TANH(c[j] + bias);
                }
            }
        }
        __syncthreads();                 // writers done; next layer may read nxt / overwrite cur
        _Float16* tmp = (_Float16*)cur; cur = nxt; nxt = tmp;
    }
    // after 3 swaps: cur == bufB holds the final h

    // ---------------- Decoder: loc+scale per thread, b128 LDS reads, float2 weights --------
    // out rows per graph (28): [j0..3, t0(2), j4..7, t1(2), j8..11, t2(2), j12..15, t3(2), j16..19]
    for (int idx = tid; idx < GPB * 28; idx += THREADS) {
        const int g   = idx / 28;
        const int row = idx % 28;
        const int seg = row / 6;
        const int w   = row % 6;
        float a0, a1;
        const _Float16* hr;
        const float2* wp;
        int wstride;                          // float2 stride between successive d
        if (w < 4) {                          // joint head: Wj[k][d][0..1] contiguous
            const int k   = 4 * seg + w;
            const int pos = 5 * seg + w + 1;
            hr = cur + (g * NPG + pos) * HD;
            a0 = bj[2 * k]; a1 = bj[2 * k + 1];
            wp = reinterpret_cast<const float2*>(Wj + k * 128);
            wstride = 1;
        } else {                              // torso head: pick (2*sub, 2*sub+1) of 4
            const int t   = seg;
            const int sub = w - 4;
            const int pos = 5 * (t + 1);
            hr = cur + (g * NPG + pos) * HD;
            a0 = bt[4 * t + 2 * sub]; a1 = bt[4 * t + 2 * sub + 1];
            wp = reinterpret_cast<const float2*>(Wt + t * 256 + 2 * sub);
            wstride = 2;
        }
#pragma unroll
        for (int dc = 0; dc < 8; ++dc) {      // 8 x ds_load_b128
            const H8 hv = *reinterpret_cast<const H8*>(hr + dc * 8);
#pragma unroll
            for (int e = 0; e < 4; ++e) {
                const int d = dc * 8 + 2 * e;
                const float h0 = (float)hv.q[e][0];
                const float h1 = (float)hv.q[e][1];
                const float2 w0 = wp[(d    ) * wstride];
                const float2 w1 = wp[(d + 1) * wstride];
                a0 += h0 * w0.x + h1 * w1.x;
                a1 += h0 * w0.y + h1 * w1.y;
            }
        }
        const int b = blockIdx.x * GPB + g;
        out[b * 28 + row] = a0;                                   // loc
        const float z = a1 + 0.5413248546129181f;                 // softplus bias log(e-1)
        out[BQ * 28 + b * 28 + row] = (z > 20.f) ? z : log1pf(__expf(z));   // scale
    }
}

extern "C" void kernel_launch(void* const* d_in, const int* in_sizes, int n_in,
                              void* d_out, int out_size, void* d_ws, size_t ws_size,
                              hipStream_t stream) {
    const float* x       = (const float*)d_in[0];
    // d_in[1] = edge_index: unused — chain topology is exploited analytically
    const float* W_torso = (const float*)d_in[2];
    const float* b_torso = (const float*)d_in[3];
    const float* W_joint = (const float*)d_in[4];
    const float* b_joint = (const float*)d_in[5];
    const float* Wrel    = (const float*)d_in[6];
    const float* brel    = (const float*)d_in[7];
    const float* Wroot   = (const float*)d_in[8];
    const float* Wj      = (const float*)d_in[9];
    const float* bj      = (const float*)d_in[10];
    const float* Wt      = (const float*)d_in[11];
    const float* bt      = (const float*)d_in[12];
    float* out = (float*)d_out;

    dim3 grid(BQ / GPB);   // 2048 blocks, 16 graphs each
    joint_actor_fused_kernel<<<grid, THREADS, 2 * HBYTES, stream>>>(
        x, W_torso, b_torso, W_joint, b_joint,
        Wrel, brel, Wroot, Wj, bj, Wt, bt, out);
}
